// NonLocalBlock2D_5669356831471
// MI455X (gfx1250) — matrix operand order repaired and checked by reading, not compiled, not run on hardware
//
#include <hip/hip_runtime.h>
#include <hip/hip_bf16.h>

#define NPIX 6400   // H*W = 80*80
#define CIN  64
#define CI   32     // inter
#define BATCH 4
#define LOG2E 1.44269504088896340736f

typedef __attribute__((ext_vector_type(16))) _Float16 v16h;
typedef __attribute__((ext_vector_type(8)))  _Float16 v8h;
typedef __attribute__((ext_vector_type(8)))  float    v8f;

// WMMA A-fragment (16x32 f16) lane slice from two contiguous 16B chunks.
// A layout as used here: lane m=l&15, half h=l>>4; elems 0..7 -> K=h*16+0..7,
// elems 8..15 -> K=h*16+8..15 (contiguous k = 16h+e, same order as the B operands).
__device__ __forceinline__ v16h make_a(const _Float16* lo_p, const _Float16* hi_p) {
  v8h lo = *(const v8h*)lo_p;
  v8h hi = *(const v8h*)hi_p;
  v16h r;
#pragma unroll
  for (int e = 0; e < 8; ++e) { r[e] = lo[e]; r[e + 8] = hi[e]; }
  return r;
}

// Exchange a value with the other 16-lane half of the wave (xor 16) via a
// single ds_swizzle_b32 (group-of-32 mode: and=0x1F, or=0, xor=0x10).
__device__ __forceinline__ float xor16(float v) {
  return __int_as_float(__builtin_amdgcn_ds_swizzle(__float_as_int(v), 0x401F));
}

// ---------------------------------------------------------------------------
// Phase 1: theta/phi/g projections.
// theta, phi stored [B, N, 32] f16 (channel contiguous)
// g stored transposed [B, 32, N] f16 (pixel contiguous per channel)
// theta is pre-scaled by log2(e) so the attention loop can use raw v_exp_f32
// (hardware exp2) with no per-element base-conversion multiply.
// ---------------------------------------------------------------------------
__global__ __launch_bounds__(256) void proj_kernel(
    const float* __restrict__ x,
    const float* __restrict__ w_theta, const float* __restrict__ b_theta,
    const float* __restrict__ w_phi,   const float* __restrict__ b_phi,
    const float* __restrict__ w_g,     const float* __restrict__ b_g,
    _Float16* __restrict__ theta, _Float16* __restrict__ phi,
    _Float16* __restrict__ gT)
{
  int tid  = blockIdx.x * 256 + threadIdx.x;   // 3*B*CI*NPIX threads
  int n    = tid % NPIX;                        // fastest: coalesced x reads
  int rest = tid / NPIX;                        // wave-uniform below
  int i    = rest & 31;
  int r2   = rest >> 5;
  int b    = r2 & 3;
  int proj = r2 >> 2;                           // 0=theta 1=phi 2=g

  const float* w;  const float* bias;
  if (proj == 0)      { w = w_theta; bias = b_theta; }
  else if (proj == 1) { w = w_phi;   bias = b_phi;   }
  else                { w = w_g;     bias = b_g;     }

  float acc = bias[i];
  const float* xr = x + (size_t)b * CIN * NPIX + n;
  const float* wr = w + i * CIN;
#pragma unroll 8
  for (int c = 0; c < CIN; ++c)
    acc += wr[c] * xr[(size_t)c * NPIX];

  if (proj == 2)
    gT[((size_t)b * CI + i) * NPIX + n] = (_Float16)acc;
  else if (proj == 0)
    theta[((size_t)(b * NPIX + n)) * CI + i] = (_Float16)(acc * LOG2E);
  else
    phi[((size_t)(b * NPIX + n)) * CI + i] = (_Float16)acc;
}

// ---------------------------------------------------------------------------
// Phase 2: fused flash-attention, transposed formulation.
// One wave per 16-query tile. S^T = phi * theta^T (keys=M, queries=N),
// softmax per query == per lane (scalar state), O^T = G * P^T (channels=M).
// Per 32-key block: 2 score WMMAs + 2 accumulate WMMAs. Scores are in the
// log2 domain (theta pre-scaled), so P = exp2(S - m) directly.
// ---------------------------------------------------------------------------
__global__ __launch_bounds__(256) void attn_kernel(
    const _Float16* __restrict__ theta, const _Float16* __restrict__ phi,
    const _Float16* __restrict__ gT, float* __restrict__ yT)
{
  const int lane = threadIdx.x & 31;
  const int wave = threadIdx.x >> 5;
  const int tile = blockIdx.x * 8 + wave;       // 0..1599
  const int b    = tile / (NPIX / 16);
  const int qb   = (tile % (NPIX / 16)) * 16;
  const int col  = lane & 15;
  const int h    = lane >> 4;

  // Loop-invariant B fragment: theta^T (32 ch x 16 queries).
  // Lane: query=qb+col, elem e -> channel h*16+e : contiguous 32B.
  const v16h BT = *(const v16h*)(theta + ((size_t)(b * NPIX + qb + col)) * CI + h * 16);

  v8f O0 = {}, O1 = {};                // O^T: channels (r+8h | +16) x queries(col)
  float m = -1.0e30f, lsum = 0.0f;     // per-query scalars (log2 domain)
  const v8f zero = {};

  const _Float16* phib = phi + (size_t)b * NPIX * CI;
  const _Float16* g0   = gT + ((size_t)b * CI + col) * NPIX;        // ch tile 0 row
  const _Float16* g1   = gT + ((size_t)b * CI + 16 + col) * NPIX;   // ch tile 1 row

#pragma unroll 2
  for (int kb = 0; kb < NPIX; kb += 32) {
    // A fragments: phi key tiles (16 keys x 32 ch).
    const _Float16* pr0 = phib + (size_t)(kb + col) * CI;
    const _Float16* pr1 = pr0 + 16 * CI;
    v16h A0 = make_a(pr0 + h * 16, pr0 + h * 16 + 8);
    v16h A1 = make_a(pr1 + h * 16, pr1 + h * 16 + 8);
    // S^T: lane l elem r -> key kb + r + 8h (S0) / kb+16+r+8h (S1), query col.
    v8f S0 = __builtin_amdgcn_wmma_f32_16x16x32_f16(false, A0, false, BT, (short)0, zero, false, false);
    v8f S1 = __builtin_amdgcn_wmma_f32_16x16x32_f16(false, A1, false, BT, (short)0, zero, false, false);

    // Per-query block max: tree over own 16 elems + one xor-16 exchange.
    float bm = fmaxf(S0[0], S1[0]);
#pragma unroll
    for (int r = 1; r < 8; ++r) bm = fmaxf(bm, fmaxf(S0[r], S1[r]));
    bm = fmaxf(bm, xor16(bm));

    float mn   = fmaxf(m, bm);
    float corr = __builtin_amdgcn_exp2f(m - mn);
    m = mn;

    // P = exp2(S - m); per-query partial sum + one xor-16 exchange.
    float e0[8], e1[8];
    float rs = 0.0f;
#pragma unroll
    for (int r = 0; r < 8; ++r) {
      e0[r] = __builtin_amdgcn_exp2f(S0[r] - m);
      e1[r] = __builtin_amdgcn_exp2f(S1[r] - m);
      rs += e0[r] + e1[r];
    }
    rs += xor16(rs);
    lsum = lsum * corr + rs;
#pragma unroll
    for (int r = 0; r < 8; ++r) { O0[r] *= corr; O1[r] *= corr; }

    // Build B fragment of P^T (32 keys x 16 queries) with a half-wave swap:
    // lane h=0 owns keys 0..7 (e0), needs 8..15 (partner e0);
    // lane h=1 owns keys 24..31 (e1), needs 16..23 (partner e1).
    v16h BP;
#pragma unroll
    for (int r = 0; r < 8; ++r) {
      float recv = xor16(h ? e0[r] : e1[r]);
      BP[r]     = (_Float16)(h ? recv : e0[r]);
      BP[r + 8] = (_Float16)(h ? e1[r] : recv);
    }

    // A fragments: G channel tiles (16 ch x 32 keys), contiguous from gT.
    v16h GA0 = make_a(g0 + kb + h * 16, g0 + kb + h * 16 + 8);
    v16h GA1 = make_a(g1 + kb + h * 16, g1 + kb + h * 16 + 8);
    O0 = __builtin_amdgcn_wmma_f32_16x16x32_f16(false, GA0, false, BP, (short)0, O0, false, false);
    O1 = __builtin_amdgcn_wmma_f32_16x16x32_f16(false, GA1, false, BP, (short)0, O1, false, false);
  }

  // Normalize and write y^T [B, 32, N] f32 (coalesced: queries contiguous).
  float inv = 1.0f / lsum;
  float* yr = yT + (size_t)b * CI * NPIX + qb + col;
#pragma unroll
  for (int r = 0; r < 8; ++r) {
    yr[(size_t)(r + 8 * h) * NPIX]      = O0[r] * inv;
    yr[(size_t)(16 + r + 8 * h) * NPIX] = O1[r] * inv;
  }
}

// ---------------------------------------------------------------------------
// Phase 3: W_y = w_out * y + b_out, batchnorm, residual. 4 outputs/thread.
// y^T is [B, 32, N] so all loads/stores are coalesced over n.
// ---------------------------------------------------------------------------
__global__ __launch_bounds__(256) void out_kernel(
    const float* __restrict__ yT, const float* __restrict__ x,
    const float* __restrict__ w_out, const float* __restrict__ b_out,
    const float* __restrict__ gamma, const float* __restrict__ beta,
    const float* __restrict__ mean,  const float* __restrict__ var,
    float* __restrict__ out)
{
  int tid  = blockIdx.x * 256 + threadIdx.x;   // B*16*NPIX threads
  int n    = tid % NPIX;
  int rest = tid / NPIX;
  int og   = rest & 15;
  int b    = rest >> 4;

  float acc[4];
#pragma unroll
  for (int j = 0; j < 4; ++j) acc[j] = b_out[og * 4 + j];
  const float* yb = yT + (size_t)b * CI * NPIX + n;
#pragma unroll
  for (int c = 0; c < CI; ++c) {
    float yv = yb[(size_t)c * NPIX];
#pragma unroll
    for (int j = 0; j < 4; ++j)
      acc[j] += w_out[(og * 4 + j) * CI + c] * yv;
  }
#pragma unroll
  for (int j = 0; j < 4; ++j) {
    int o = og * 4 + j;
    float inv = gamma[o] * __frsqrt_rn(var[o] + 1e-4f);
    size_t idx = ((size_t)(b * CIN) + o) * NPIX + n;
    out[idx] = (acc[j] - mean[o]) * inv + beta[o] + x[idx];
  }
}

// ---------------------------------------------------------------------------
extern "C" void kernel_launch(void* const* d_in, const int* in_sizes, int n_in,
                              void* d_out, int out_size, void* d_ws, size_t ws_size,
                              hipStream_t stream)
{
  const float* x       = (const float*)d_in[0];
  const float* w_theta = (const float*)d_in[1];
  const float* b_theta = (const float*)d_in[2];
  const float* w_phi   = (const float*)d_in[3];
  const float* b_phi   = (const float*)d_in[4];
  const float* w_g     = (const float*)d_in[5];
  const float* b_g     = (const float*)d_in[6];
  const float* w_out   = (const float*)d_in[7];
  const float* b_out   = (const float*)d_in[8];
  const float* gamma   = (const float*)d_in[9];
  const float* beta    = (const float*)d_in[10];
  const float* mean    = (const float*)d_in[11];
  const float* var     = (const float*)d_in[12];
  float* out = (float*)d_out;

  char* ws = (char*)d_ws;
  const size_t elems = (size_t)BATCH * NPIX * CI;      // 819200
  _Float16* theta = (_Float16*)ws;
  _Float16* phi   = theta + elems;
  _Float16* gT    = phi + elems;
  float*    yT    = (float*)(ws + 3 * elems * sizeof(_Float16));

  proj_kernel<<<(3 * BATCH * CI * NPIX) / 256, 256, 0, stream>>>(
      x, w_theta, b_theta, w_phi, b_phi, w_g, b_g, theta, phi, gT);
  attn_kernel<<<(BATCH * (NPIX / 16)) / 8, 256, 0, stream>>>(theta, phi, gT, yT);
  out_kernel<<<(BATCH * 16 * NPIX) / 256, 256, 0, stream>>>(
      yT, x, w_out, b_out, gamma, beta, mean, var, out);
}
